// DeepNeuralDecisionForest_28046136443194
// MI455X (gfx1250) — matrix operand order repaired and checked by reading, not compiled, not run on hardware
//
#include <hip/hip_runtime.h>
#include <math.h>

typedef __attribute__((ext_vector_type(16))) _Float16 v16h;
typedef __attribute__((ext_vector_type(8)))  _Float16 v8h;
typedef __attribute__((ext_vector_type(8)))  float    v8f;

constexpr int kT  = 16;    // trees
constexpr int kB  = 1024;  // batch
constexpr int kF  = 512;   // features
constexpr int kH  = 128;   // hidden
constexpr int kL  = 512;   // leaves (2^(D+1), D=8)
constexpr int kC  = 100;   // classes
constexpr int kCp = 112;   // classes padded to multiple of 16

// ---------------------------------------------------------------------------
// WMMA fragment helpers (wave32, v_wmma_f32_16x16x32_f16)
//
// A (16x32 f16): lane holds row M = lane&15; kh = lane>>4 selects K sub-block.
//   element j<8  -> K = kh*8 + j        (8 contiguous halves at kk + kh*8)
//   element j>=8 -> K = 16 + kh*8 + j-8 (8 contiguous halves at kk + 16 + kh*8)
// B (32x16 f16): lane holds col N = lane&15; same K mapping as A
//   (8 halves at kk + kh*8 and 8 at kk + 16 + kh*8, B stored K-contiguous per col)
// C/D (16x16 f32): element r -> row M = r + 8*kh, col N = lane&15
// ---------------------------------------------------------------------------
__device__ __forceinline__ v16h load_a_frag(const _Float16* rowbase, int kk, int kh) {
  const _Float16* p = rowbase + kk + kh * 8;
  v8h lo = *(const v8h*)(p);
  v8h hi = *(const v8h*)(p + 16);
  return __builtin_shufflevector(lo, hi, 0,1,2,3,4,5,6,7,8,9,10,11,12,13,14,15);
}

__device__ __forceinline__ v16h load_b_frag(const _Float16* colbase, int kk, int kh) {
  const _Float16* p = colbase + kk + kh * 8;
  v8h lo = *(const v8h*)(p);
  v8h hi = *(const v8h*)(p + 16);
  return __builtin_shufflevector(lo, hi, 0,1,2,3,4,5,6,7,8,9,10,11,12,13,14,15);
}

__device__ __forceinline__ v8f wmma_f16(v16h a, v16h b, v8f c) {
  // 8 args: (neg_a, A, neg_b, B, c_mod, C, reuse_a, reuse_b)
  return __builtin_amdgcn_wmma_f32_16x16x32_f16(false, a, false, b, (short)0, c,
                                                false, false);
}

// ---------------------------------------------------------------------------
// Repack / convert kernels
// ---------------------------------------------------------------------------
__global__ __launch_bounds__(256) void cvt_feat_kernel(const float* __restrict__ src,
                                                       _Float16* __restrict__ dst, int n) {
  int i = blockIdx.x * 256 + threadIdx.x;
  if (i < n) dst[i] = (_Float16)src[i];
}

// W1 (t,f,h) f32 -> W1T (t,h,f) f16  (B operand: K=f contiguous per column h)
__global__ __launch_bounds__(256) void repack_w1_kernel(const float* __restrict__ W1,
                                                        _Float16* __restrict__ W1T) {
  int i = blockIdx.x * 256 + threadIdx.x;      // over kT*kH*kF
  int t = i / (kH * kF);
  int r = i % (kH * kF);
  int n = r / kF;
  int k = r % kF;
  W1T[i] = (_Float16)W1[((size_t)t * kF + k) * kH + n];
}

// W2 (t,h,l) f32 -> W2T (t,l,h) f16  (B operand: K=h contiguous per column l)
__global__ __launch_bounds__(256) void repack_w2_kernel(const float* __restrict__ W2,
                                                        _Float16* __restrict__ W2T) {
  int i = blockIdx.x * 256 + threadIdx.x;      // over kT*kL*kH
  int t = i / (kL * kH);
  int r = i % (kL * kH);
  int l = r / kH;
  int k = r % kH;
  W2T[i] = (_Float16)W2[((size_t)t * kH + k) * kL + l];
}

// leaf_p = softmax(pi, axis=-1); store transposed+padded: LPT[t][c(0..111)][l]
// One wave per (t,l) row of 100 classes.
__global__ __launch_bounds__(256) void softmax_pi_kernel(const float* __restrict__ pi,
                                                         _Float16* __restrict__ LPT) {
  int row  = blockIdx.x * 8 + (threadIdx.x >> 5);   // t*kL + l
  int lane = threadIdx.x & 31;
  int t = row / kL, l = row % kL;
  const float* src = pi + (size_t)row * kC;

  float v[4];
  float mx = -1e30f;
  #pragma unroll
  for (int j = 0; j < 4; ++j) {
    int c = lane + 32 * j;
    v[j] = (c < kC) ? src[c] : -1e30f;
    mx = fmaxf(mx, v[j]);
  }
  #pragma unroll
  for (int s = 16; s > 0; s >>= 1) mx = fmaxf(mx, __shfl_xor(mx, s, 32));

  float sum = 0.f;
  #pragma unroll
  for (int j = 0; j < 4; ++j) {
    int c = lane + 32 * j;
    v[j] = (c < kC) ? __expf(v[j] - mx) : 0.f;
    sum += v[j];
  }
  #pragma unroll
  for (int s = 16; s > 0; s >>= 1) sum += __shfl_xor(sum, s, 32);
  float inv = 1.0f / sum;

  #pragma unroll
  for (int j = 0; j < 4; ++j) {
    int c = lane + 32 * j;
    if (c < kCp) {
      _Float16 o = (c < kC) ? (_Float16)(v[j] * inv) : (_Float16)0.0f;
      LPT[((size_t)t * kCp + c) * kL + l] = o;
    }
  }
}

// ---------------------------------------------------------------------------
// GEMM1: h[t] = relu(feat @ W1[t] + b1[t]) ; M=1024 N=128 K=512, f16 out
// One wave computes a 16x64 strip (4 N-tiles) reusing the A fragment.
// All fragment loads hoisted ahead of the 4 WMMAs so the backend can overlap
// loads with matrix ops (partial s_wait_loadcnt instead of wait-all).
// waves = 16 * 64 * 2 = 2048
// ---------------------------------------------------------------------------
__global__ __launch_bounds__(256) void gemm1_kernel(const _Float16* __restrict__ featH,
                                                    const _Float16* __restrict__ W1T,
                                                    const float* __restrict__ b1,
                                                    _Float16* __restrict__ hH) {
  int gw   = (blockIdx.x * 256 + threadIdx.x) >> 5;
  int lane = threadIdx.x & 31;
  int kh = lane >> 4, lr = lane & 15;
  int ng = gw & 1;                 // which 64-wide N strip
  int mt = (gw >> 1) & 63;
  int t  = gw >> 7;
  int b0 = mt * 16, n0 = ng * 64;

  const _Float16* Arow  = featH + (size_t)(b0 + lr) * kF;
  const _Float16* Bbase = W1T + ((size_t)t * kH + (n0 + lr)) * kF;

  v8f acc0 = {}, acc1 = {}, acc2 = {}, acc3 = {};
  #pragma unroll 2
  for (int kk = 0; kk < kF; kk += 32) {
    v16h a   = load_a_frag(Arow, kk, kh);
    v16h bf0 = load_b_frag(Bbase,           kk, kh);
    v16h bf1 = load_b_frag(Bbase + 16 * kF, kk, kh);
    v16h bf2 = load_b_frag(Bbase + 32 * kF, kk, kh);
    v16h bf3 = load_b_frag(Bbase + 48 * kF, kk, kh);
    acc0 = wmma_f16(a, bf0, acc0);
    acc1 = wmma_f16(a, bf1, acc1);
    acc2 = wmma_f16(a, bf2, acc2);
    acc3 = wmma_f16(a, bf3, acc3);
  }

  v8f accs[4] = {acc0, acc1, acc2, acc3};
  #pragma unroll
  for (int j = 0; j < 4; ++j) {
    int n = n0 + j * 16 + lr;
    float bias = b1[t * kH + n];
    #pragma unroll
    for (int r = 0; r < 8; ++r) {
      float x = accs[j][r] + bias;
      x = x > 0.f ? x : 0.f;                     // relu
      hH[((size_t)t * kB + (b0 + r + 8 * kh)) * kH + n] = (_Float16)x;
    }
  }
}

// ---------------------------------------------------------------------------
// GEMM2: p[t] = sigmoid(h[t] @ W2[t] + b2[t]) ; M=1024 N=512 K=128, f32 out
// One wave computes a 16x64 strip (4 N-tiles). waves = 16 * 64 * 8 = 8192
// ---------------------------------------------------------------------------
__global__ __launch_bounds__(256) void gemm2_kernel(const _Float16* __restrict__ hH,
                                                    const _Float16* __restrict__ W2T,
                                                    const float* __restrict__ b2,
                                                    float* __restrict__ pBuf) {
  int gw   = (blockIdx.x * 256 + threadIdx.x) >> 5;
  int lane = threadIdx.x & 31;
  int kh = lane >> 4, lr = lane & 15;
  int ng = gw & 7;                 // which 64-wide N strip (512/64 = 8)
  int mt = (gw >> 3) & 63;
  int t  = gw >> 9;
  int b0 = mt * 16, n0 = ng * 64;

  const _Float16* Arow  = hH + ((size_t)t * kB + b0 + lr) * kH;
  const _Float16* Bbase = W2T + ((size_t)t * kL + (n0 + lr)) * kH;

  v8f acc0 = {}, acc1 = {}, acc2 = {}, acc3 = {};
  #pragma unroll
  for (int kk = 0; kk < kH; kk += 32) {
    v16h a   = load_a_frag(Arow, kk, kh);
    v16h bf0 = load_b_frag(Bbase,           kk, kh);
    v16h bf1 = load_b_frag(Bbase + 16 * kH, kk, kh);
    v16h bf2 = load_b_frag(Bbase + 32 * kH, kk, kh);
    v16h bf3 = load_b_frag(Bbase + 48 * kH, kk, kh);
    acc0 = wmma_f16(a, bf0, acc0);
    acc1 = wmma_f16(a, bf1, acc1);
    acc2 = wmma_f16(a, bf2, acc2);
    acc3 = wmma_f16(a, bf3, acc3);
  }

  v8f accs[4] = {acc0, acc1, acc2, acc3};
  #pragma unroll
  for (int j = 0; j < 4; ++j) {
    int n = n0 + j * 16 + lr;
    float bias = b2[t * kL + n];
    #pragma unroll
    for (int r = 0; r < 8; ++r) {
      float x = accs[j][r] + bias;
      float s = 1.0f / (1.0f + __expf(-x));      // sigmoid
      pBuf[((size_t)t * kB + (b0 + r + 8 * kh)) * kL + n] = s;
    }
  }
}

// ---------------------------------------------------------------------------
// Routing: mu[t,b,l] = prod_{d=0..8} (side ? 1-p[node] : p[node])
// One block per (t,b); p row (512 f32) cached in LDS.
// ---------------------------------------------------------------------------
__global__ __launch_bounds__(512) void route_kernel(const float* __restrict__ pBuf,
                                                    _Float16* __restrict__ muH) {
  __shared__ float srow[kL];
  int tb = blockIdx.x;                 // t*kB + b
  int l  = threadIdx.x;
  srow[l] = pBuf[(size_t)tb * kL + l];
  __syncthreads();
  float m = 1.0f;
  #pragma unroll
  for (int d = 0; d < 9; ++d) {
    int node = (1 << d) - 1 + (l >> (9 - d));
    int side = (l >> (8 - d)) & 1;
    float pv = srow[node];
    m *= side ? (1.0f - pv) : pv;
  }
  muH[(size_t)tb * kL + l] = (_Float16)m;
}

// ---------------------------------------------------------------------------
// GEMM3: out[b,c] = log( (1/(L*T)) * sum_{t,l} mu[t,b,l] * leafp[t,l,c] )
// K fused over t (K_total = 8192); M=1024, N=112 (store only c<100)
// One wave per 16x16 tile -> 448 waves of parallelism.
// ---------------------------------------------------------------------------
__global__ __launch_bounds__(256) void gemm3_kernel(const _Float16* __restrict__ muH,
                                                    const _Float16* __restrict__ LPT,
                                                    float* __restrict__ out) {
  int gw   = (blockIdx.x * 256 + threadIdx.x) >> 5;
  int lane = threadIdx.x & 31;
  int kh = lane >> 4, lr = lane & 15;
  int nt = gw % 7;
  int mt = gw / 7;
  int b0 = mt * 16, c0 = nt * 16;

  v8f acc = {};
  for (int t = 0; t < kT; ++t) {
    const _Float16* Arow = muH + ((size_t)t * kB + b0 + lr) * kL;
    const _Float16* Brow = LPT + ((size_t)t * kCp + (c0 + lr)) * kL;
    #pragma unroll 4
    for (int kk = 0; kk < kL; kk += 32)
      acc = wmma_f16(load_a_frag(Arow, kk, kh), load_b_frag(Brow, kk, kh), acc);
  }

  const float inv = 1.0f / (float)(kL * kT);
  int c = c0 + lr;
  if (c < kC) {
    #pragma unroll
    for (int r = 0; r < 8; ++r)
      out[(size_t)(b0 + r + 8 * kh) * kC + c] = logf(acc[r] * inv);
  }
}

// ---------------------------------------------------------------------------
// Host launcher
// ---------------------------------------------------------------------------
extern "C" void kernel_launch(void* const* d_in, const int* in_sizes, int n_in,
                              void* d_out, int out_size, void* d_ws, size_t ws_size,
                              hipStream_t stream) {
  const float* feat = (const float*)d_in[0];   // (B,F)
  const float* W1   = (const float*)d_in[1];   // (T,F,H)
  const float* b1   = (const float*)d_in[2];   // (T,H)
  const float* W2   = (const float*)d_in[3];   // (T,H,L)
  const float* b2   = (const float*)d_in[4];   // (T,L)
  const float* pi   = (const float*)d_in[5];   // (T,L,C)
  float* out = (float*)d_out;                  // (B,C) log-probs

  char* ws = (char*)d_ws;
  size_t off = 0;
  auto take = [&](size_t bytes) -> char* {
    char* p = ws + off;
    off = (off + bytes + 255) & ~(size_t)255;
    return p;
  };
  _Float16* featH = (_Float16*)take((size_t)kB * kF * 2);        //  1.0 MB
  _Float16* W1T   = (_Float16*)take((size_t)kT * kH * kF * 2);   //  2.0 MB
  _Float16* W2T   = (_Float16*)take((size_t)kT * kL * kH * 2);   //  2.0 MB
  _Float16* hH    = (_Float16*)take((size_t)kT * kB * kH * 2);   //  4.0 MB
  float*    pBuf  = (float*)   take((size_t)kT * kB * kL * 4);   // 32.0 MB
  _Float16* muH   = (_Float16*)take((size_t)kT * kB * kL * 2);   // 16.0 MB
  _Float16* LPT   = (_Float16*)take((size_t)kT * kCp * kL * 2);  //  1.8 MB

  // repack / convert
  cvt_feat_kernel<<<(kB * kF + 255) / 256, 256, 0, stream>>>(feat, featH, kB * kF);
  repack_w1_kernel<<<(kT * kH * kF) / 256, 256, 0, stream>>>(W1, W1T);
  repack_w2_kernel<<<(kT * kL * kH) / 256, 256, 0, stream>>>(W2, W2T);
  softmax_pi_kernel<<<(kT * kL) / 8, 256, 0, stream>>>(pi, LPT);

  // pipeline
  gemm1_kernel<<<(16 * 64 * 2) / 8, 256, 0, stream>>>(featH, W1T, b1, hH);   //  256 blocks
  gemm2_kernel<<<(16 * 64 * 8) / 8, 256, 0, stream>>>(hH, W2T, b2, pBuf);    // 1024 blocks
  route_kernel<<<kT * kB, 512, 0, stream>>>(pBuf, muH);                       // 16384 blocks
  gemm3_kernel<<<(64 * 7) / 8, 256, 0, stream>>>(muH, LPT, out);              //   56 blocks
}
